// ContrastSense_without_queue_23656679867748
// MI455X (gfx1250) — matrix operand order repaired and checked by reading, not compiled, not run on hardware
//
#include <hip/hip_runtime.h>
#include <hip/hip_bf16.h>

// ---------------------------------------------------------------------------
// ContrastSense (no queue) on MI455X / gfx1250.
//  inputs : q[N,C] f32, k[N,C] f32, domain_label[N] i32, time_label[N] i32
//  outputs: logits[N,N+1] f32 | targets[N] (zeros) | feature[N,N] f32
// ---------------------------------------------------------------------------

#define N_ROWS   4096
#define C_DIM    256
#define N_DOM    8
#define TIME_WIN 50
#define TEMP_F   0.07f
#define LAST_RATIO_F 0.7f

typedef __bf16 bf16;
typedef __attribute__((ext_vector_type(8)))  __bf16 v8bf;
typedef __attribute__((ext_vector_type(16))) __bf16 v16bf;
typedef __attribute__((ext_vector_type(8)))  float  v8f;

// ---------------------------------------------------------------------------
// Kernel 0: per-domain keep-thresholds (floor(count_d * 0.7)); zero targets.
// ---------------------------------------------------------------------------
__global__ void k_setup(const int* __restrict__ dom,
                        float* __restrict__ out_targets,
                        int* __restrict__ thresh) {
  __shared__ int cnt[N_DOM];
  const int t = threadIdx.x;
  if (t < N_DOM) cnt[t] = 0;
  __syncthreads();
  for (int i = t; i < N_ROWS; i += 256) atomicAdd(&cnt[dom[i]], 1);
  __syncthreads();
  if (t < N_DOM) thresh[t] = (int)floorf((float)cnt[t] * LAST_RATIO_F);
  for (int i = t; i < N_ROWS; i += 256) out_targets[i] = 0.0f;
}

// ---------------------------------------------------------------------------
// Kernel 1: normalize q,k rows; bf16 hi/lo split; l_pos -> logits[:,0]
// One wave (32 lanes) per row; each lane owns 8 contiguous features.
// ---------------------------------------------------------------------------
__global__ void k_normalize(const float* __restrict__ q,
                            const float* __restrict__ k,
                            bf16* __restrict__ qhi, bf16* __restrict__ qlo,
                            bf16* __restrict__ khi, bf16* __restrict__ klo,
                            float* __restrict__ logits) {
  const int wave = threadIdx.x >> 5;
  const int lane = threadIdx.x & 31;
  const int row  = blockIdx.x * 8 + wave;

  const float4* qv4 = (const float4*)(q + (size_t)row * C_DIM) + lane * 2;
  const float4* kv4 = (const float4*)(k + (size_t)row * C_DIM) + lane * 2;
  float4 q0 = qv4[0], q1 = qv4[1];
  float4 k0 = kv4[0], k1 = kv4[1];
  float qv[8] = {q0.x, q0.y, q0.z, q0.w, q1.x, q1.y, q1.z, q1.w};
  float kv[8] = {k0.x, k0.y, k0.z, k0.w, k1.x, k1.y, k1.z, k1.w};

  float sq = 0.f, sk = 0.f;
#pragma unroll
  for (int e = 0; e < 8; ++e) { sq += qv[e] * qv[e]; sk += kv[e] * kv[e]; }
#pragma unroll
  for (int o = 16; o >= 1; o >>= 1) {
    sq += __shfl_xor(sq, o, 32);
    sk += __shfl_xor(sk, o, 32);
  }
  const float invq = 1.0f / fmaxf(sqrtf(sq), 1e-12f);
  const float invk = 1.0f / fmaxf(sqrtf(sk), 1e-12f);

  const size_t base = (size_t)row * C_DIM + lane * 8;
  float lpos = 0.f;
#pragma unroll
  for (int e = 0; e < 8; ++e) {
    const float qn = qv[e] * invq;
    const float kn = kv[e] * invk;
    lpos += qn * kn;
    bf16 qh = (bf16)qn;  qhi[base + e] = qh;  qlo[base + e] = (bf16)(qn - (float)qh);
    bf16 kh = (bf16)kn;  khi[base + e] = kh;  klo[base + e] = (bf16)(kn - (float)kh);
  }
#pragma unroll
  for (int o = 16; o >= 1; o >>= 1) lpos += __shfl_xor(lpos, o, 32);
  if (lane == 0) logits[(size_t)row * (N_ROWS + 1)] = lpos / TEMP_F;
}

// ---------------------------------------------------------------------------
// Kernel 2: feature = qn @ kn^T via v_wmma_f32_16x16x32_bf16, bf16x2 split.
// Block = 8 waves; block tile 128(M) x 64(N); each wave: 16x64 (4 wmma tiles).
// A/B fragments use contiguous k order: lane half h holds k = 16h..16h+15 for both.
// ---------------------------------------------------------------------------
__global__ void k_gemm(const bf16* __restrict__ qhi, const bf16* __restrict__ qlo,
                       const bf16* __restrict__ khi, const bf16* __restrict__ klo,
                       float* __restrict__ feat) {
  const int lane = threadIdx.x & 31;
  const int wave = threadIdx.x >> 5;
  const int m       = lane & 15;
  const int rowBase = blockIdx.y * 128 + wave * 16;
  const int colBase = blockIdx.x * 64;
  const int koffA   = (lane < 16) ? 0 : 16;
  const int koffB   = (lane < 16) ? 0 : 16;

  v8f acc[4];
#pragma unroll
  for (int ct = 0; ct < 4; ++ct) acc[ct] = (v8f)(0.0f);

  const size_t aRow = (size_t)(rowBase + m) * C_DIM;

  for (int kb = 0; kb < C_DIM; kb += 32) {
    const bf16* aph = qhi + aRow + kb + koffA;
    const bf16* apl = qlo + aRow + kb + koffA;
    v8bf ah0 = *(const v8bf*)(aph);
    v8bf ah1 = *(const v8bf*)(aph + 8);
    v8bf al0 = *(const v8bf*)(apl);
    v8bf al1 = *(const v8bf*)(apl + 8);
    v16bf ahi = __builtin_shufflevector(ah0, ah1, 0,1,2,3,4,5,6,7,8,9,10,11,12,13,14,15);
    v16bf alo = __builtin_shufflevector(al0, al1, 0,1,2,3,4,5,6,7,8,9,10,11,12,13,14,15);

#pragma unroll
    for (int ct = 0; ct < 4; ++ct) {
      const size_t bRow = (size_t)(colBase + ct * 16 + m) * C_DIM + kb + koffB;
      v16bf bhi = *(const v16bf*)(khi + bRow);
      v16bf blo = *(const v16bf*)(klo + bRow);
      // hi*hi + hi*lo + lo*hi  ~ fp32 product (lo*lo ~ 2^-32, dropped)
      acc[ct] = __builtin_amdgcn_wmma_f32_16x16x32_bf16(false, ahi, false, bhi,
                                                        (short)0, acc[ct], false, false);
      acc[ct] = __builtin_amdgcn_wmma_f32_16x16x32_bf16(false, ahi, false, blo,
                                                        (short)0, acc[ct], false, false);
      acc[ct] = __builtin_amdgcn_wmma_f32_16x16x32_bf16(false, alo, false, bhi,
                                                        (short)0, acc[ct], false, false);
    }
  }

  const int rowHalf = (lane < 16) ? 0 : 8;
#pragma unroll
  for (int ct = 0; ct < 4; ++ct) {
#pragma unroll
    for (int r = 0; r < 8; ++r) {
      feat[(size_t)(rowBase + r + rowHalf) * N_ROWS + colBase + ct * 16 + m] = acc[ct][r];
    }
  }
}

// ---------------------------------------------------------------------------
// Kernel 3: exact per-(row,domain) rank-thresh similarity via 4-pass radix
// select (8 bits/pass, MSB first) on an order-preserving uint32 encoding.
// One block (256 threads) per row.  Replaces the O(N log^2 N) bitonic sort:
// ~20K LDS ops/row instead of ~640K, 13 barriers instead of 78.
// ---------------------------------------------------------------------------
__global__ void k_select(const float* __restrict__ feat,
                         const int* __restrict__ dom,
                         const int* __restrict__ thresh,
                         float* __restrict__ thrOut) {
  __shared__ unsigned      enc[N_ROWS];       // 16 KB
  __shared__ unsigned char doms[N_ROWS];      //  4 KB
  __shared__ int           hist[N_DOM][256];  //  8 KB
  __shared__ unsigned      chosen[N_DOM];
  __shared__ unsigned      maskSoFar[N_DOM];
  __shared__ int           want[N_DOM];

  const int row = blockIdx.x;
  const int t   = threadIdx.x;

  if (t < N_DOM) { chosen[t] = 0u; maskSoFar[t] = 0u; want[t] = thresh[t]; }
  for (int j = t; j < N_ROWS; j += 256) {
    const unsigned b = __float_as_uint(feat[(size_t)row * N_ROWS + j]);
    enc[j]  = (b & 0x80000000u) ? ~b : (b | 0x80000000u);  // monotonic in value
    doms[j] = (unsigned char)dom[j];
  }
  __syncthreads();

#pragma unroll
  for (int p = 0; p < 4; ++p) {
    const int shift = 24 - 8 * p;
    for (int i = t; i < N_DOM * 256; i += 256) ((int*)hist)[i] = 0;
    __syncthreads();
    for (int j = t; j < N_ROWS; j += 256) {
      const int d = doms[j];
      const unsigned e = enc[j];
      if ((e & maskSoFar[d]) == chosen[d])
        atomicAdd(&hist[d][(e >> shift) & 255], 1);
    }
    __syncthreads();
    if (t < N_DOM && want[t] > 0) {
      const int w = want[t];
      int cum = 0;
      unsigned b = 0;
      for (int bb = 255; bb >= 0; --bb) {
        const int c = hist[t][bb];
        if (cum + c >= w) { b = (unsigned)bb; break; }
        cum += c;
      }
      want[t]      = w - cum;            // rank within chosen bucket
      chosen[t]   |= b << shift;
      maskSoFar[t] |= 0xFFu << shift;
    }
    __syncthreads();
  }

  if (t < N_DOM) {
    float v;
    if (thresh[t] > 0) {
      const unsigned e = chosen[t];   // encoding of the last-kept similarity
      const unsigned b = (e & 0x80000000u) ? (e & 0x7FFFFFFFu) : ~e;
      v = __uint_as_float(b);         // eliminate sims strictly below this
    } else {
      v = 3.0e38f;                    // keep nothing in this domain
    }
    thrOut[(size_t)row * N_DOM + t] = v;
  }
}

// ---------------------------------------------------------------------------
// Kernel 4: masked logits:  -inf if (sim < domain_thresh) or |dt| < window,
// else sim / TEMP.  grid = (4, N); each thread handles 4 consecutive cols.
// ---------------------------------------------------------------------------
__global__ void k_logits(const float* __restrict__ feat,
                         const int* __restrict__ dom,
                         const int* __restrict__ tim,
                         const float* __restrict__ thr,
                         float* __restrict__ logits) {
  __shared__ float th[N_DOM];
  __shared__ int   ti_s;
  const int row = blockIdx.y;
  const int t   = threadIdx.x;
  if (t < N_DOM) th[t] = thr[(size_t)row * N_DOM + t];
  if (t == 0)    ti_s  = tim[row];
  __syncthreads();

  const int ti = ti_s;
  const int j0 = blockIdx.x * 1024 + t * 4;
  float4 s4 = *(const float4*)(feat + (size_t)row * N_ROWS + j0);
  const float sv[4] = {s4.x, s4.y, s4.z, s4.w};
  float* outp = logits + (size_t)row * (N_ROWS + 1) + 1 + j0;

#pragma unroll
  for (int e = 0; e < 4; ++e) {
    const int j = j0 + e;
    const int d = dom[j];
    int dt = tim[j] - ti; dt = (dt < 0) ? -dt : dt;
    const bool elim = (sv[e] < th[d]) | (dt < TIME_WIN);
    outp[e] = elim ? -__builtin_inff() : (sv[e] / TEMP_F);  // row stride N+1: scalar stores
  }
}

// ---------------------------------------------------------------------------
extern "C" void kernel_launch(void* const* d_in, const int* in_sizes, int n_in,
                              void* d_out, int out_size, void* d_ws, size_t ws_size,
                              hipStream_t stream) {
  const float* q   = (const float*)d_in[0];
  const float* k   = (const float*)d_in[1];
  const int*   dom = (const int*)d_in[2];
  const int*   tim = (const int*)d_in[3];

  float* out      = (float*)d_out;
  float* logits   = out;                                             // [N, N+1]
  float* targets  = out + (size_t)N_ROWS * (N_ROWS + 1);             // [N]
  float* feature  = targets + N_ROWS;                                // [N, N]

  char* w = (char*)d_ws;
  const size_t HALF = (size_t)N_ROWS * C_DIM * sizeof(bf16);         // 2 MB
  bf16* qhi = (bf16*)(w);
  bf16* qlo = (bf16*)(w + HALF);
  bf16* khi = (bf16*)(w + 2 * HALF);
  bf16* klo = (bf16*)(w + 3 * HALF);
  float* thrv = (float*)(w + 4 * HALF);                              // [N, 8]
  int* thresh = (int*)(w + 4 * HALF + (size_t)N_ROWS * N_DOM * sizeof(float));

  k_setup<<<1, 256, 0, stream>>>(dom, targets, thresh);
  k_normalize<<<N_ROWS / 8, 256, 0, stream>>>(q, k, qhi, qlo, khi, klo, logits);
  k_gemm<<<dim3(N_ROWS / 64, N_ROWS / 128), 256, 0, stream>>>(qhi, qlo, khi, klo, feature);
  k_select<<<N_ROWS, 256, 0, stream>>>(feature, dom, thresh, thrv);
  k_logits<<<dim3(4, N_ROWS), 256, 0, stream>>>(feature, dom, tim, thrv, logits);
}
